// DoubleRNNPOSAwareLM_18141941858877
// MI455X (gfx1250) — matrix operand order repaired and checked by reading, not compiled, not run on hardware
//
#include <hip/hip_runtime.h>
#include <hip/hip_bf16.h>

// ---------------------------------------------------------------------------
// CDNA5 (gfx1250) implementation of the POS-aware double-LSTM LM.
//   * bf16 WMMA (v_wmma_f32_16x16x32_bf16) for all GEMMs, fp32 accumulate.
//   * Recurrent weights converted fp32->bf16 once per call (~70 MB, resident
//     in the 192 MB L2 across the 128 sequential steps).
//   * LSTM-cell kernels stage the shared A-operand (activations) into LDS via
//     GLOBAL_LOAD_ASYNC_TO_LDS_B128 (ASYNCcnt-tracked) and read A-fragments
//     from LDS; weights stream global->VGPR (single use, prefetched to GL2).
//   * Output projections via a generic WMMA GEMM + log-softmax kernels.
// ---------------------------------------------------------------------------

typedef __bf16  bf16;
typedef __bf16  v16bf __attribute__((ext_vector_type(16)));
typedef float   v8f   __attribute__((ext_vector_type(8)));
typedef int     v4i   __attribute__((ext_vector_type(4)));

#define WMMA_BF16(a, b, c) \
    __builtin_amdgcn_wmma_f32_16x16x32_bf16(false, (a), false, (b), (short)0, (c), false, false)

// Async global->LDS copy support (gfx1250): guarded so compile never breaks.
#if defined(__has_builtin)
#  if __has_builtin(__builtin_amdgcn_global_load_async_to_lds_b128) && \
      __has_builtin(__builtin_amdgcn_s_wait_asynccnt)
#    define USE_ASYNC_LDS 1
#  endif
#endif
#ifndef USE_ASYNC_LDS
#  define USE_ASYNC_LDS 0
#endif

#if USE_ASYNC_LDS
typedef __attribute__((address_space(1))) v4i* global_v4i_p;   // "__device__" AS
typedef __attribute__((address_space(3))) v4i* lds_v4i_p;      // "__shared__" AS
#endif

// ---- constants ----
#define T_STEPS 128
#define BATCH   32
#define POS_V   45
#define WORD_V  32000
#define POS_E   64
#define WORD_E  512
#define POS_H   256
#define WORD_H  1024
#define TB      (T_STEPS * BATCH)   // 4096
#define LDS_PAD 8                   // halves; makes A-frag LDS reads conflict-free

// ---------------------------------------------------------------------------
// Fragment loaders; A and B share one contiguous k order (k = 16*(lane/16) + s).
// A (16x32 bf16):  lanes 0-15 hold M=lane, K = k0..k0+15 (contiguous);
//                  lanes 16-31 hold K = k0+16..k0+31.
// B (32x16 bf16):  lanes 0-15 hold N=lane, K = k0..k0+15 (contiguous);
//                  lanes 16-31 hold K = k0+16..k0+31.
// Per-lane 16-byte-aligned contiguous chunks -> global_load_b128/ds_load_b128.
// ---------------------------------------------------------------------------
__device__ __forceinline__ v16bf load_a_frag(const bf16* x, int ldx,
                                             int row0, int k0, int lane) {
    int row = row0 + (lane & 15);
    int sel = (lane >> 4) << 4;                 // 0 or 16
    const bf16* p = x + (size_t)row * ldx + k0 + sel;
    v16bf a;
    reinterpret_cast<v4i*>(&a)[0] = *reinterpret_cast<const v4i*>(p);        // K: k0+sel..+7
    reinterpret_cast<v4i*>(&a)[1] = *reinterpret_cast<const v4i*>(p + 8);    // K: k0+sel+8..+15
    return a;
}

__device__ __forceinline__ v16bf load_b_frag(const bf16* __restrict__ w, int ldw,
                                             int col, int k0, int lane) {
    int c   = col + (lane & 15);                // output column == weight row (W is [N,K])
    int sel = (lane >> 4) << 4;                 // 0 or 16
    const bf16* p = w + (size_t)c * ldw + k0 + sel;
    v16bf b;
    reinterpret_cast<v4i*>(&b)[0] = reinterpret_cast<const v4i*>(p)[0];
    reinterpret_cast<v4i*>(&b)[1] = reinterpret_cast<const v4i*>(p)[1];
    return b;
}

// Accumulate one K-segment of gates: acc[2][4] covers a 32(M) x 64(N) tile.
// All 4 B-fragments are materialized before the 8 WMMAs so the scheduler can
// clause the loads and issue a single wait per K-step.
__device__ __forceinline__ void seg_accumulate(v8f (&acc)[2][4],
                                               const bf16* x, int ldx,
                                               const bf16* __restrict__ w, int ldw,
                                               int K, int colBase, int lane) {
    for (int k0 = 0; k0 < K; k0 += 32) {
        if (k0 + 32 < K)  // stream next weight tile toward GL2 (global_prefetch_b8)
            __builtin_prefetch(w + (size_t)(colBase + (lane & 15)) * ldw + k0 + 32, 0, 0);
        v16bf a0 = load_a_frag(x, ldx, 0,  k0, lane);
        v16bf a1 = load_a_frag(x, ldx, 16, k0, lane);
        v16bf b0 = load_b_frag(w, ldw, colBase +  0, k0, lane);
        v16bf b1 = load_b_frag(w, ldw, colBase + 16, k0, lane);
        v16bf b2 = load_b_frag(w, ldw, colBase + 32, k0, lane);
        v16bf b3 = load_b_frag(w, ldw, colBase + 48, k0, lane);
        acc[0][0] = WMMA_BF16(a0, b0, acc[0][0]);
        acc[1][0] = WMMA_BF16(a1, b0, acc[1][0]);
        acc[0][1] = WMMA_BF16(a0, b1, acc[0][1]);
        acc[1][1] = WMMA_BF16(a1, b1, acc[1][1]);
        acc[0][2] = WMMA_BF16(a0, b2, acc[0][2]);
        acc[1][2] = WMMA_BF16(a1, b2, acc[1][2]);
        acc[0][3] = WMMA_BF16(a0, b3, acc[0][3]);
        acc[1][3] = WMMA_BF16(a1, b3, acc[1][3]);
    }
}

// Stage one x-segment ([BATCH, K] bf16, row-major) into padded LDS image
// ([BATCH, ldl] at column offset coff), in 128-bit chunks.
__device__ __forceinline__ void stage_x(bf16* xlds, int ldl, int coff,
                                        const bf16* __restrict__ x, int K, int tid) {
    const int cpr   = K >> 3;           // 16-byte chunks per row
    const int total = BATCH * cpr;
    for (int i = tid; i < total; i += 128) {
        int row = i / cpr;
        int kc  = (i - row * cpr) << 3;
        const bf16* g = x    + (size_t)row * K   + kc;
        bf16*       l = xlds + (size_t)row * ldl + coff + kc;
#if USE_ASYNC_LDS
        __builtin_amdgcn_global_load_async_to_lds_b128(
            (global_v4i_p)(v4i*)g, (lds_v4i_p)(v4i*)l, 0, 0);
#else
        *reinterpret_cast<v4i*>(l) = *reinterpret_cast<const v4i*>(g);
#endif
    }
}

// ---------------------------------------------------------------------------
// LSTM cell kernel.  grid.x = H/64, block = 128 threads (4 waves).
// Wave g computes gate-block g (i,f,g,o) for 64 hidden columns; gates are
// exchanged via LDS and the elementwise LSTM update is applied in-kernel.
// Up to 3 K-segments: gate = x0@W0^T + x1@W1^T + x2@W2^T + bias.
// Dynamic LDS: [4][32][64] f32 gate buffer, then padded x image [32][Ktot+8].
// ---------------------------------------------------------------------------
__global__ __launch_bounds__(128)
void lstm_cell(const bf16* __restrict__ x0, const bf16* __restrict__ w0, int K0, int ldw0,
               const bf16* __restrict__ x1, const bf16* __restrict__ w1, int K1, int ldw1,
               const bf16* __restrict__ x2, const bf16* __restrict__ w2, int K2, int ldw2,
               const float* __restrict__ bias, int H,
               float* __restrict__ c_io, bf16* __restrict__ h_out) {
    extern __shared__ char smem[];
    float (*gbuf)[BATCH][64] = (float (*)[BATCH][64])smem;               // 32 KB
    bf16* xlds = (bf16*)(smem + 4 * BATCH * 64 * sizeof(float));

    const int tid  = threadIdx.x;
    const int wave = tid >> 5;                  // gate index: 0=i 1=f 2=g 3=o
    const int lane = tid & 31;
    const int j0   = blockIdx.x * 64;           // hidden-column slab
    const int colBase = wave * H + j0;          // row offset into [4H,K] weights

    const int Ktot = K0 + K1 + K2;
    const int ldl  = Ktot + LDS_PAD;

    // Stage all x segments into LDS (async when available).
    stage_x(xlds, ldl, 0, x0, K0, tid);
    if (K1) stage_x(xlds, ldl, K0, x1, K1, tid);
    if (K2) stage_x(xlds, ldl, K0 + K1, x2, K2, tid);
#if USE_ASYNC_LDS
    __builtin_amdgcn_s_wait_asynccnt(0);
#endif
    __syncthreads();

    v8f acc[2][4] = {};
    seg_accumulate(acc, xlds, ldl, w0, ldw0, K0, colBase, lane);
    if (K1) seg_accumulate(acc, xlds + K0, ldl, w1, ldw1, K1, colBase, lane);
    if (K2) seg_accumulate(acc, xlds + K0 + K1, ldl, w2, ldw2, K2, colBase, lane);

    // Scatter accumulators to LDS (C/D layout: lanes 0-15 M=r, lanes 16-31 M=r+8).
    const int nrow = (lane >> 4) << 3;
#pragma unroll
    for (int mt = 0; mt < 2; ++mt)
#pragma unroll
        for (int nt = 0; nt < 4; ++nt)
#pragma unroll
            for (int r = 0; r < 8; ++r)
                gbuf[wave][16 * mt + nrow + r][16 * nt + (lane & 15)] = acc[mt][nt][r];
    __syncthreads();

    // Elementwise LSTM update: 32*64 = 2048 elements over 128 threads.
#pragma unroll
    for (int e = 0; e < 16; ++e) {
        int f  = e * 128 + tid;
        int b  = f >> 6;
        int jl = f & 63;
        int j  = j0 + jl;
        float gi = gbuf[0][b][jl] + bias[j];
        float gf = gbuf[1][b][jl] + bias[H + j];
        float gg = gbuf[2][b][jl] + bias[2 * H + j];
        float go = gbuf[3][b][jl] + bias[3 * H + j];
        float si = 1.f / (1.f + __expf(-gi));
        float sf = 1.f / (1.f + __expf(-gf));
        float so = 1.f / (1.f + __expf(-go));
        float c  = c_io[b * H + j];
        float c2 = sf * c + si * tanhf(gg);
        float h2 = so * tanhf(c2);
        c_io[b * H + j]  = c2;
        h_out[b * H + j] = (bf16)h2;
    }
}

// ---------------------------------------------------------------------------
// Generic WMMA GEMM: C[M,N] = A[M,K] @ W[N,K]^T + bias.
// Block = 128 threads (4 waves), block tile 32(M) x 256(N); wave tile 32x64.
// Writes fp32 (Cf) or bf16 (Cb).
// ---------------------------------------------------------------------------
__global__ __launch_bounds__(128)
void gemm_bf16(const bf16* __restrict__ A, int lda,
               const bf16* __restrict__ W, int ldw,
               const float* __restrict__ bias,
               int K,
               float* __restrict__ Cf, bf16* __restrict__ Cb, int ldc) {
    const int wave = threadIdx.x >> 5;
    const int lane = threadIdx.x & 31;
    const int row0 = blockIdx.y * 32;
    const int col0 = blockIdx.x * 256 + wave * 64;
    const bf16* Ab = A + (size_t)row0 * lda;

    v8f acc[2][4] = {};
    seg_accumulate(acc, Ab, lda, W, ldw, K, col0, lane);

    const int nrow = (lane >> 4) << 3;
#pragma unroll
    for (int mt = 0; mt < 2; ++mt)
#pragma unroll
        for (int nt = 0; nt < 4; ++nt) {
            int col  = col0 + 16 * nt + (lane & 15);
            float bv = bias ? bias[col] : 0.f;
#pragma unroll
            for (int r = 0; r < 8; ++r) {
                int row = row0 + 16 * mt + nrow + r;
                float v = acc[mt][nt][r] + bv;
                if (Cf) Cf[(size_t)row * ldc + col] = v;
                else    Cb[(size_t)row * ldc + col] = (bf16)v;
            }
        }
}

// ---------------------------------------------------------------------------
// Utility kernels
// ---------------------------------------------------------------------------
__global__ void cvt_f32_bf16(const float* __restrict__ s, bf16* __restrict__ d, long n) {
    long i = (long)blockIdx.x * blockDim.x + threadIdx.x;
    long st = (long)gridDim.x * blockDim.x;
    for (; i < n; i += st) d[i] = (bf16)s[i];
}

__global__ void bias_combine(const float* __restrict__ a, const float* __restrict__ b,
                             float* __restrict__ o, int n) {
    int i = blockIdx.x * blockDim.x + threadIdx.x;
    if (i < n) o[i] = a[i] + b[i];
}

__global__ void zero_f32(float* p, long n) {
    long i = (long)blockIdx.x * blockDim.x + threadIdx.x;
    long st = (long)gridDim.x * blockDim.x;
    for (; i < n; i += st) p[i] = 0.f;
}

__global__ void zero_bf16(bf16* p, long n) {
    long i = (long)blockIdx.x * blockDim.x + threadIdx.x;
    long st = (long)gridDim.x * blockDim.x;
    for (; i < n; i += st) p[i] = (bf16)0.f;
}

// gather rows from an f32 embedding table -> bf16 sequence buffer
__global__ void gather_emb(const int* __restrict__ idx, const float* __restrict__ table,
                           int E, bf16* __restrict__ out, long total) {
    long i = (long)blockIdx.x * blockDim.x + threadIdx.x;
    long st = (long)gridDim.x * blockDim.x;
    for (; i < total; i += st) {
        long tb = i / E;
        int  c  = (int)(i - tb * E);
        out[i] = (bf16)table[(size_t)idx[tb] * E + c];
    }
}

// pos projection (V=45, K=256): plain VALU dot products + log-softmax.
__global__ __launch_bounds__(64)
void pos_proj_lsm(const bf16* __restrict__ ph,      // flat [TB,256]
                  const float* __restrict__ Wp,     // [45,256]
                  const float* __restrict__ bp,     // [45]
                  float* __restrict__ out) {        // [TB,45]
    __shared__ float hrow[POS_H];
    __shared__ float lg[POS_V];
    __shared__ float stat[2];
    const int r = blockIdx.x, tid = threadIdx.x;
    for (int k = tid; k < POS_H; k += 64) hrow[k] = (float)ph[(size_t)r * POS_H + k];
    __syncthreads();
    if (tid < POS_V) {
        float s = bp[tid];
        const float* wr = Wp + tid * POS_H;
        for (int k = 0; k < POS_H; ++k) s += hrow[k] * wr[k];
        lg[tid] = s;
    }
    __syncthreads();
    if (tid == 0) {
        float m = lg[0];
        for (int j = 1; j < POS_V; ++j) m = fmaxf(m, lg[j]);
        float s = 0.f;
        for (int j = 0; j < POS_V; ++j) s += __expf(lg[j] - m);
        stat[0] = m; stat[1] = __logf(s);
    }
    __syncthreads();
    if (tid < POS_V) out[(size_t)r * POS_V + tid] = lg[tid] - stat[0] - stat[1];
}

// in-place row log-softmax for the big vocab (n = 32000)
__global__ __launch_bounds__(256)
void logsoftmax_rows(float* __restrict__ x, int n) {
    __shared__ float red[256];
    const int r = blockIdx.x, tid = threadIdx.x;
    float* row = x + (size_t)r * n;
    float m = -3.4e38f;
    for (int i = tid; i < n; i += 256) m = fmaxf(m, row[i]);
    red[tid] = m; __syncthreads();
    for (int s = 128; s > 0; s >>= 1) { if (tid < s) red[tid] = fmaxf(red[tid], red[tid + s]); __syncthreads(); }
    m = red[0]; __syncthreads();
    float sum = 0.f;
    for (int i = tid; i < n; i += 256) sum += __expf(row[i] - m);
    red[tid] = sum; __syncthreads();
    for (int s = 128; s > 0; s >>= 1) { if (tid < s) red[tid] += red[tid + s]; __syncthreads(); }
    const float lz = m + __logf(red[0]);
    for (int i = tid; i < n; i += 256) row[i] -= lz;
}

// ---------------------------------------------------------------------------
// Host launcher
// ---------------------------------------------------------------------------
extern "C" void kernel_launch(void* const* d_in, const int* in_sizes, int n_in,
                              void* d_out, int out_size, void* d_ws, size_t ws_size,
                              hipStream_t stream) {
    (void)in_sizes; (void)n_in; (void)out_size; (void)ws_size;

    const int*   pos        = (const int*)  d_in[0];
    const int*   word       = (const int*)  d_in[1];
    const float* pos_emb_W  = (const float*)d_in[2];
    const float* word_emb_W = (const float*)d_in[3];
    const float* pos_Wih    = (const float*)d_in[4];
    const float* pos_Whh    = (const float*)d_in[5];
    const float* pos_bih    = (const float*)d_in[6];
    const float* pos_bhh    = (const float*)d_in[7];
    const float* w0_Wih     = (const float*)d_in[8];
    const float* w0_Whh     = (const float*)d_in[9];
    const float* w0_bih     = (const float*)d_in[10];
    const float* w0_bhh     = (const float*)d_in[11];
    const float* w1_Wih     = (const float*)d_in[12];
    const float* w1_Whh     = (const float*)d_in[13];
    const float* w1_bih     = (const float*)d_in[14];
    const float* w1_bhh     = (const float*)d_in[15];
    const float* pos_proj_W = (const float*)d_in[16];
    const float* pos_proj_b = (const float*)d_in[17];
    const float* wp1_W      = (const float*)d_in[18];
    const float* wp1_b      = (const float*)d_in[19];
    const float* wp2_b      = (const float*)d_in[20];

    // --- workspace bump allocator (256B aligned) ---
    char* base = (char*)d_ws;
    size_t off = 0;
    auto alloc = [&](size_t bytes) -> void* {
        void* p = base + off;
        off = (off + bytes + 255) & ~(size_t)255;
        return p;
    };
    bf16* posWihB = (bf16*)alloc((size_t)1024 * 1088 * 2);   // [4*POS_H, POS_E+WORD_H]
    bf16* posWhhB = (bf16*)alloc((size_t)1024 * 256  * 2);
    bf16* w0WihB  = (bf16*)alloc((size_t)4096 * 768  * 2);   // [4*WORD_H, WORD_E+POS_H]
    bf16* w0WhhB  = (bf16*)alloc((size_t)4096 * 1024 * 2);
    bf16* w1WihB  = (bf16*)alloc((size_t)4096 * 1024 * 2);
    bf16* w1WhhB  = (bf16*)alloc((size_t)4096 * 1024 * 2);
    bf16* wp1WB   = (bf16*)alloc((size_t)512  * 1024 * 2);
    bf16* wembB   = (bf16*)alloc((size_t)WORD_V * WORD_E * 2);
    bf16* pembS   = (bf16*)alloc((size_t)TB * POS_E  * 2);
    bf16* wembS   = (bf16*)alloc((size_t)TB * WORD_E * 2);
    bf16* phH     = (bf16*)alloc((size_t)(T_STEPS + 1) * BATCH * POS_H  * 2);
    bf16* wh1H    = (bf16*)alloc((size_t)(T_STEPS + 1) * BATCH * WORD_H * 2);
    bf16* wh0D    = (bf16*)alloc((size_t)2 * BATCH * WORD_H * 2);
    bf16* wmid    = (bf16*)alloc((size_t)TB * WORD_E * 2);
    float* biasP  = (float*)alloc(1024 * 4);
    float* biasW0 = (float*)alloc(4096 * 4);
    float* biasW1 = (float*)alloc(4096 * 4);
    float* pc     = (float*)alloc((size_t)BATCH * POS_H  * 4);
    float* w0c    = (float*)alloc((size_t)BATCH * WORD_H * 4);
    float* w1c    = (float*)alloc((size_t)BATCH * WORD_H * 4);

    auto cvt = [&](const float* s, bf16* d, long n) {
        int g = (int)((n + 1023) / 1024);
        cvt_f32_bf16<<<g, 256, 0, stream>>>(s, d, n);
    };
    cvt(pos_Wih, posWihB, 1024L * 1088);
    cvt(pos_Whh, posWhhB, 1024L * 256);
    cvt(w0_Wih,  w0WihB,  4096L * 768);
    cvt(w0_Whh,  w0WhhB,  4096L * 1024);
    cvt(w1_Wih,  w1WihB,  4096L * 1024);
    cvt(w1_Whh,  w1WhhB,  4096L * 1024);
    cvt(wp1_W,   wp1WB,   512L  * 1024);
    cvt(word_emb_W, wembB, (long)WORD_V * WORD_E);

    bias_combine<<<4, 256, 0, stream>>>(pos_bih, pos_bhh, biasP, 1024);
    bias_combine<<<16, 256, 0, stream>>>(w0_bih, w0_bhh, biasW0, 4096);
    bias_combine<<<16, 256, 0, stream>>>(w1_bih, w1_bhh, biasW1, 4096);

    gather_emb<<<256, 256, 0, stream>>>(pos,  pos_emb_W,  POS_E,  pembS, (long)TB * POS_E);
    gather_emb<<<2048, 256, 0, stream>>>(word, word_emb_W, WORD_E, wembS, (long)TB * WORD_E);

    // zero initial states (h0 slots and cell states) every call
    zero_bf16<<<8,  256, 0, stream>>>(phH,  (long)BATCH * POS_H);
    zero_bf16<<<32, 256, 0, stream>>>(wh1H, (long)BATCH * WORD_H);
    zero_bf16<<<32, 256, 0, stream>>>(wh0D, (long)BATCH * WORD_H);
    zero_f32 <<<8,  256, 0, stream>>>(pc,   (long)BATCH * POS_H);
    zero_f32 <<<32, 256, 0, stream>>>(w0c,  (long)BATCH * WORD_H);
    zero_f32 <<<32, 256, 0, stream>>>(w1c,  (long)BATCH * WORD_H);

    // dynamic LDS: 32KB gate buffer + padded x image [32][Ktot+8] bf16
    auto cell_shmem = [](int Ktot) -> size_t {
        return (size_t)4 * BATCH * 64 * sizeof(float)
             + (size_t)BATCH * (Ktot + LDS_PAD) * sizeof(bf16);
    };
    const size_t shA = cell_shmem(POS_E + WORD_H + POS_H);   // 1344
    const size_t shB = cell_shmem(WORD_E + POS_H + WORD_H);  // 1792
    const size_t shC = cell_shmem(WORD_H + WORD_H);          // 2048

    // --- sequential recurrence: 3 dependent cells per timestep ---
    for (int t = 0; t < T_STEPS; ++t) {
        // pos cell: gates = [p_emb ; wh1(t)] @ pos_Wih^T + ph(t) @ pos_Whh^T
        lstm_cell<<<POS_H / 64, 128, shA, stream>>>(
            pembS + (size_t)t * BATCH * POS_E,  posWihB,        POS_E,  1088,
            wh1H  + (size_t)t * BATCH * WORD_H, posWihB + POS_E, WORD_H, 1088,
            phH   + (size_t)t * BATCH * POS_H,  posWhhB,        POS_H,  POS_H,
            biasP, POS_H, pc,
            phH + (size_t)(t + 1) * BATCH * POS_H);
        // word layer 0: gates = [w_emb ; ph(t+1)] @ w0_Wih^T + wh0_prev @ w0_Whh^T
        lstm_cell<<<WORD_H / 64, 128, shB, stream>>>(
            wembS + (size_t)t * BATCH * WORD_E,       w0WihB,          WORD_E, 768,
            phH   + (size_t)(t + 1) * BATCH * POS_H,  w0WihB + WORD_E, POS_H,  768,
            wh0D  + (size_t)(t & 1) * BATCH * WORD_H, w0WhhB,          WORD_H, WORD_H,
            biasW0, WORD_H, w0c,
            wh0D + (size_t)((t + 1) & 1) * BATCH * WORD_H);
        // word layer 1: gates = wh0_new @ w1_Wih^T + wh1(t) @ w1_Whh^T
        lstm_cell<<<WORD_H / 64, 128, shC, stream>>>(
            wh0D + (size_t)((t + 1) & 1) * BATCH * WORD_H, w1WihB, WORD_H, WORD_H,
            wh1H + (size_t)t * BATCH * WORD_H,             w1WhhB, WORD_H, WORD_H,
            nullptr, nullptr, 0, 1,
            biasW1, WORD_H, w1c,
            wh1H + (size_t)(t + 1) * BATCH * WORD_H);
    }

    // --- projections ---
    float* p_lp = (float*)d_out;                       // [TB, 45]
    float* w_lp = p_lp + (size_t)TB * POS_V;           // [TB, 32000]

    pos_proj_lsm<<<TB, 64, 0, stream>>>(phH + (size_t)BATCH * POS_H,
                                        pos_proj_W, pos_proj_b, p_lp);

    // w_mid[TB,512] = wh1 @ wp1_W^T + wp1_b   (bf16 out)
    {
        dim3 g(WORD_E / 256, TB / 32);
        gemm_bf16<<<g, 128, 0, stream>>>(wh1H + (size_t)BATCH * WORD_H, WORD_H,
                                         wp1WB, WORD_H, wp1_b, WORD_H,
                                         nullptr, wmid, WORD_E);
    }
    // logits[TB,32000] = w_mid @ word_emb^T + wp2_b   (f32 out to d_out)
    {
        dim3 g(WORD_V / 256, TB / 32);
        gemm_bf16<<<g, 128, 0, stream>>>(wmid, WORD_E,
                                         wembB, WORD_E, wp2_b, WORD_E,
                                         w_lp, nullptr, WORD_V);
    }
    logsoftmax_rows<<<TB, 256, 0, stream>>>(w_lp, WORD_V);
}
